// MSDeformAttn_60567628808640
// MI455X (gfx1250) — matrix operand order repaired and checked by reading, not compiled, not run on hardware
//
#include <hip/hip_runtime.h>
#include <hip/hip_bf16.h>

// ---------------------------------------------------------------------------
// MSDeformAttn forward for MI455X (gfx1250, wave32, WMMA)
//   query (2,21760,256) f32 ; values (16,32,HW_l) f32 ; out (2,21760,256) f32
// ---------------------------------------------------------------------------

typedef __attribute__((ext_vector_type(16))) __bf16 v16bf;
typedef __attribute__((ext_vector_type(16))) float  v16f;
typedef __attribute__((ext_vector_type(8)))  float  v8f;

#define NBATCH  2
#define LQ      21760
#define NROWS   (NBATCH * LQ)        // 43520
#define DMODEL  256
#define NHEADS  8
#define DHEAD   32
#define NCOLS   384                  // 256 offset cols + 128 attn cols
#define KTILES  8                    // K = 256 = 8 * 32
#define NTILES  (NCOLS / 16)         // 24
#define MTILES  (NROWS / 16)         // 2720
#define NT_PER_WAVE 4                // N-tiles per wave in the GEMM

// ---------------------------------------------------------------------------
// Kernel 1: pack B = [W_off | W_attn] (f32, KxN row-major) into bf16 WMMA
// B-fragments. K order used here (contiguous, same as A) for v_wmma_*_16x16x32 B (32x16, K x N):
//   lane holds column N = lane%16; per-lane element j -> K = j + 16*(lane>=16)
// Storage: Bfrag[((ntile*8 + kk)*32 + lane)*16 + j]
// ---------------------------------------------------------------------------
__global__ __launch_bounds__(256)
void prep_bfrag(const float* __restrict__ Woff,
                const float* __restrict__ Wattn,
                __bf16* __restrict__ Bfrag) {
    int t = blockIdx.x * blockDim.x + threadIdx.x;     // 0 .. 24*8*32-1
    if (t >= NTILES * KTILES * 32) return;
    int lane  = t & 31;
    int kk    = (t >> 5) & (KTILES - 1);
    int ntile = t >> 8;
    int col   = ntile * 16 + (lane & 15);
    int kbase = kk * 32 + ((lane >> 4) << 4);
    v16f w;
#pragma unroll
    for (int j = 0; j < 16; ++j) {
        int k = kbase + j;
        w[j] = (col < 256) ? Woff[(size_t)k * 256 + col]
                           : Wattn[(size_t)k * 128 + (col - 256)];
    }
    *(v16bf*)(Bfrag + (size_t)t * 16) = __builtin_convertvector(w, v16bf);
}

// ---------------------------------------------------------------------------
// Kernel 2: logits = query @ [W_off|W_attn] + bias, via bf16 WMMA.
// One wave computes a 16x64 strip (1 M-tile x 4 N-tiles), reusing the
// converted A fragment across 4 WMMAs. 8 waves/block -> block = 128x64 tile.
// A fragment (16x32, 16-bit): row = lane%16, K base = 16*(lane>=16),
//   elements j -> K=base+j (matches Bfrag packing).
// ---------------------------------------------------------------------------
__global__ __launch_bounds__(256)
void gemm_wmma(const float* __restrict__ A,
               const __bf16* __restrict__ Bfrag,
               const float* __restrict__ boff,
               const float* __restrict__ battn,
               float* __restrict__ logits) {
    const int lane  = threadIdx.x & 31;
    const int wave  = threadIdx.x >> 5;
    const int mtile = blockIdx.x * 8 + wave;
    const int nt0   = blockIdx.y * NT_PER_WAVE;
    const int row   = mtile * 16 + (lane & 15);
    const int kb    = (lane >> 4) * 16;

    v8f acc[NT_PER_WAVE];
#pragma unroll
    for (int i = 0; i < NT_PER_WAVE; ++i)
        acc[i] = (v8f){0.f, 0.f, 0.f, 0.f, 0.f, 0.f, 0.f, 0.f};

    const float* ap = A + (size_t)row * DMODEL + kb;
    const __bf16* bp = Bfrag + ((size_t)(nt0 * KTILES) * 32 + lane) * 16;

#pragma unroll
    for (int kk = 0; kk < KTILES; ++kk) {
        const float4* a4 = (const float4*)(ap + kk * 32);
        float4 a0 = a4[0];           // K = base+0..3
        float4 a1 = a4[1];           // K = base+4..7
        float4 a2 = a4[2];           // K = base+8..11
        float4 a3 = a4[3];           // K = base+12..15
        v16f fa;
        fa[0]  = a0.x; fa[1]  = a0.y; fa[2]  = a0.z; fa[3]  = a0.w;
        fa[4]  = a1.x; fa[5]  = a1.y; fa[6]  = a1.z; fa[7]  = a1.w;
        fa[8]  = a2.x; fa[9]  = a2.y; fa[10] = a2.z; fa[11] = a2.w;
        fa[12] = a3.x; fa[13] = a3.y; fa[14] = a3.z; fa[15] = a3.w;
        v16bf va = __builtin_convertvector(fa, v16bf);   // native f32->bf16

#pragma unroll
        for (int i = 0; i < NT_PER_WAVE; ++i) {
            v16bf vb = *(const v16bf*)(bp + ((size_t)i * KTILES + kk) * 32 * 16);
            acc[i] = __builtin_amdgcn_wmma_f32_16x16x32_bf16(
                         false, va, false, vb, (short)0, acc[i], false, false);
        }
    }

    const int r0 = mtile * 16 + ((lane >> 4) << 3);   // M = r (lo lanes) / 8+r (hi)
#pragma unroll
    for (int i = 0; i < NT_PER_WAVE; ++i) {
        const int col  = (nt0 + i) * 16 + (lane & 15);
        const float bias = (col < 256) ? boff[col] : battn[col - 256];
#pragma unroll
        for (int r = 0; r < 8; ++r)
            logits[(size_t)(r0 + r) * NCOLS + col] = acc[i][r] + bias;
    }
}

// ---------------------------------------------------------------------------
// Kernel 3: transpose one value level (B',32,HW) -> (B',HW,32) so the 32
// channels of a spatial sample sit in a single 128B cache line.
// ---------------------------------------------------------------------------
__global__ __launch_bounds__(256)
void transpose_vals(const float* __restrict__ v, float* __restrict__ t, int HW) {
    int tid = blockIdx.x * blockDim.x + threadIdx.x;  // = (bp*HW + sidx)*32 + d
    int total = NBATCH * NHEADS * DHEAD * HW;
    if (tid >= total) return;
    int d    = tid & 31;
    int rest = tid >> 5;
    int sidx = rest % HW;
    int bp   = rest / HW;
    t[tid] = v[((size_t)bp * DHEAD + d) * HW + sidx];
}

// ---------------------------------------------------------------------------
// Kernel 4: softmax + bilinear sampling + weighted sum.
// One wave per (n, q, head); lane = channel d (0..31). Location/softmax math
// is wave-uniform (broadcast loads); corner gathers are coalesced when useT.
// ---------------------------------------------------------------------------
__global__ __launch_bounds__(256)
void msda_sample(const float* __restrict__ logits,
                 const float* __restrict__ refp,
                 const float* __restrict__ v0, const float* __restrict__ v1,
                 const float* __restrict__ v2, const float* __restrict__ v3,
                 float* __restrict__ out, int useT) {
    const int lane = threadIdx.x & 31;
    const int wid  = (blockIdx.x * blockDim.x + threadIdx.x) >> 5;
    const int m    = wid & 7;
    const int row  = wid >> 3;                 // n*Lq + q
    if (row >= NROWS) return;

    const float* lrow = logits + (size_t)row * NCOLS;

    // softmax over the 16 (level,point) attention logits of this head
    float lg[16];
    float mx = -1e30f;
#pragma unroll
    for (int i = 0; i < 16; ++i) {
        lg[i] = lrow[256 + m * 16 + i];
        mx = fmaxf(mx, lg[i]);
    }
    float sum = 0.f;
#pragma unroll
    for (int i = 0; i < 16; ++i) {
        lg[i] = __expf(lg[i] - mx);
        sum += lg[i];
    }
    const float inv = 1.f / sum;

    const int n   = row / LQ;
    const int bpi = n * NHEADS + m;
    const float* vls[4] = {v0, v1, v2, v3};
    const int Hs[4] = {128, 64, 32, 16};

    float acc = 0.f;
#pragma unroll
    for (int l = 0; l < 4; ++l) {
        const int H = Hs[l], W = Hs[l], HW = H * W;
        const float rx = refp[((size_t)row * 4 + l) * 2 + 0];
        const float ry = refp[((size_t)row * 4 + l) * 2 + 1];
        const float* vl = vls[l];
#pragma unroll
        for (int p = 0; p < 4; ++p) {
            const int oc = ((m * 4 + l) * 4 + p) * 2;
            const float ox = lrow[oc + 0];
            const float oy = lrow[oc + 1];
            // (ref + off/norm)*dim - 0.5  ==  ref*dim + off - 0.5
            const float x = rx * (float)W + ox - 0.5f;
            const float y = ry * (float)H + oy - 0.5f;
            const float aw = lg[l * 4 + p] * inv;

            const float x0f = floorf(x), y0f = floorf(y);
            const int x0 = (int)x0f, y0 = (int)y0f;
            const float wx = x - x0f, wy = y - y0f;

            float s = 0.f;
#pragma unroll
            for (int cy = 0; cy < 2; ++cy) {
                const int yi = y0 + cy;
                if (yi < 0 || yi >= H) continue;
                const float wyv = cy ? wy : (1.f - wy);
#pragma unroll
                for (int cx = 0; cx < 2; ++cx) {
                    const int xi = x0 + cx;
                    if (xi < 0 || xi >= W) continue;
                    const float wxv = cx ? wx : (1.f - wx);
                    const int si = yi * W + xi;
                    const float val =
                        useT ? vl[((size_t)bpi * HW + si) * DHEAD + lane]
                             : vl[((size_t)bpi * DHEAD + lane) * HW + si];
                    s += wxv * wyv * val;
                }
            }
            acc += aw * s;
        }
    }
    out[(size_t)row * DMODEL + m * DHEAD + lane] = acc;
}

// ---------------------------------------------------------------------------
extern "C" void kernel_launch(void* const* d_in, const int* in_sizes, int n_in,
                              void* d_out, int out_size, void* d_ws, size_t ws_size,
                              hipStream_t stream) {
    (void)in_sizes; (void)n_in; (void)out_size;

    const float* query = (const float*)d_in[0];
    const float* refp  = (const float*)d_in[1];
    const float* v[4]  = {(const float*)d_in[2], (const float*)d_in[3],
                          (const float*)d_in[4], (const float*)d_in[5]};
    const float* Woff  = (const float*)d_in[6];
    const float* boff  = (const float*)d_in[7];
    const float* Wattn = (const float*)d_in[8];
    const float* battn = (const float*)d_in[9];
    float* out = (float*)d_out;

    const size_t logitsBytes = (size_t)NROWS * NCOLS * sizeof(float);
    const size_t bfragBytes  = (size_t)NTILES * KTILES * 32 * 16 * sizeof(__bf16);
    const size_t bfragAl     = (bfragBytes + 255) & ~(size_t)255;

    const int HWs[4] = {16384, 4096, 1024, 256};
    size_t tOff[4], tot = 0;
    for (int l = 0; l < 4; ++l) {
        tOff[l] = tot;
        tot += (size_t)NBATCH * NHEADS * DHEAD * HWs[l] * sizeof(float);
    }
    const size_t needT = logitsBytes + bfragAl + tot;

    char* ws = (char*)d_ws;
    float*  logits = (float*)ws;
    __bf16* bfrag  = (__bf16*)(ws + logitsBytes);
    char*   tbase  = ws + logitsBytes + bfragAl;
    const int useT = (ws_size >= needT) ? 1 : 0;

    // 1) pack weights into bf16 WMMA fragments
    prep_bfrag<<<dim3((NTILES * KTILES * 32 + 255) / 256), dim3(256), 0, stream>>>(
        Woff, Wattn, bfrag);

    // 2) fused projection GEMM (offsets + attn logits) via v_wmma
    gemm_wmma<<<dim3(MTILES / 8, NTILES / NT_PER_WAVE), dim3(256), 0, stream>>>(
        query, bfrag, boff, battn, logits);

    // 3) optional value transpose for coalesced channel gathers
    const float* sv[4];
    if (useT) {
        for (int l = 0; l < 4; ++l) {
            float* t = (float*)(tbase + tOff[l]);
            int total = NBATCH * NHEADS * DHEAD * HWs[l];
            transpose_vals<<<dim3(total / 256), dim3(256), 0, stream>>>(v[l], t, HWs[l]);
            sv[l] = t;
        }
    } else {
        for (int l = 0; l < 4; ++l) sv[l] = v[l];
    }

    // 4) softmax + bilinear sampling + weighted reduce
    msda_sample<<<dim3(NROWS * NHEADS / 8), dim3(256), 0, stream>>>(
        logits, refp, sv[0], sv[1], sv[2], sv[3], out, useT);
}
